// LSTM_54262616818002
// MI455X (gfx1250) — matrix operand order repaired and checked by reading, not compiled, not run on hardware
//
#include <hip/hip_runtime.h>

// ---------------------------------------------------------------------------
// Problem dims
// ---------------------------------------------------------------------------
#define H_DIM 512
#define B_DIM 32768
#define KTOT  1536            // 512 (input) + 512 (out0/rec) + 512 (mem0)
#define NK    48              // KTOT / 32 k-steps

// LDS tile geometry (bf16, 32 k per step). Each 64B row gets 16B TDM pad
// -> 80B (20 dword) row stride, conflict-free ds_load_b128 fragment reads.
#define ROWB      80
#define X_ROWS    128                     // batch rows per WG tile
#define W_ROWS    256                     // 4 gates * 64 h-rows
#define X_BYTES   (X_ROWS * ROWB)         // 10240
#define W_BYTES   (W_ROWS * ROWB)         // 20480
#define BUF_BYTES (X_BYTES + W_BYTES)     // 30720 (x2 buffers = 60KB LDS)

typedef __bf16        v16bf __attribute__((ext_vector_type(16)));
typedef float         v8f   __attribute__((ext_vector_type(8)));
typedef unsigned int  u32x4 __attribute__((ext_vector_type(4)));
typedef int           i32x8 __attribute__((ext_vector_type(8)));
typedef int           i32x4 __attribute__((ext_vector_type(4)));

// ---------------------------------------------------------------------------
// TDM descriptor helpers (CDNA5 Tensor Data Mover, D# groups per ISA 08 §8)
// ---------------------------------------------------------------------------
// group1 dw0: data_size=2B (1<<16) | pad_enable (1<<20) | pad_interval=16DW
// (3<<22) | pad_amount=4DW (3<<25)
__device__ inline i32x8 make_g1(unsigned td0, unsigned td1,
                                unsigned tl0, unsigned tl1, unsigned tl2,
                                unsigned long long s0, unsigned long long s1)
{
    i32x8 g;
    g[0] = (int)((1u << 16) | (1u << 20) | (3u << 22) | (3u << 25));
    g[1] = (int)((td0 & 0xFFFFu) << 16);                                   // tensor_dim0 lo
    g[2] = (int)(((td0 >> 16) & 0xFFFFu) | ((td1 & 0xFFFFu) << 16));       // dim0 hi | dim1 lo
    g[3] = (int)(((td1 >> 16) & 0xFFFFu) | ((tl0 & 0xFFFFu) << 16));       // dim1 hi | tile0
    g[4] = (int)((tl1 & 0xFFFFu) | ((tl2 & 0xFFFFu) << 16));               // tile1 | tile2
    g[5] = (int)(s0 & 0xFFFFFFFFull);                                      // stride0 lo32
    g[6] = (int)(((s0 >> 32) & 0xFFFFull) | ((s1 & 0xFFFFull) << 16));     // s0 hi16 | s1 lo16
    g[7] = (int)((s1 >> 16) & 0xFFFFFFFFull);                              // s1 [47:16]
    return g;
}

__device__ inline void tdm_load(unsigned lds_addr, unsigned long long ga,
                                i32x8 g1, i32x4 g2)
{
    u32x4 g0;
    g0[0] = 1u;                                   // count=1, user descriptor
    g0[1] = lds_addr;                             // LDS byte address
    g0[2] = (unsigned)ga;                         // global addr [31:0]
    g0[3] = (unsigned)((ga >> 32) & 0x01FFFFFFull) | 0x80000000u; // [56:32] | type=2
    i32x4 g3 = {0, 0, 0, 0};
    i32x8 gx = {0, 0, 0, 0, 0, 0, 0, 0};          // 6-arg form: extra group (unused)
    __builtin_amdgcn_tensor_load_to_lds(g0, g1, g2, g3, gx, 0);
}

// ---------------------------------------------------------------------------
// Kernel 1: pack 11 weight matrices into W_packed[2048][1536] bf16 + bias[2048]
// row r = gate*512 + h ; cols [0,512)=W_x, [512,1024)=W_rec, [1024,1536)=W_mem
// gate order: 0=block_inp(tanh), 1=inp_gate, 2=read_gate, 3=write_gate
// ---------------------------------------------------------------------------
__global__ void pack_weights(
    const float* __restrict__ wi0, const float* __restrict__ wr0,
    const float* __restrict__ wi1, const float* __restrict__ wr1, const float* __restrict__ wm1,
    const float* __restrict__ wi2, const float* __restrict__ wr2, const float* __restrict__ wm2,
    const float* __restrict__ wi3, const float* __restrict__ wr3, const float* __restrict__ wm3,
    __bf16* __restrict__ Wp, float* __restrict__ bias)
{
    int k = blockIdx.x * blockDim.x + threadIdx.x;   // 0..1535
    int r = blockIdx.y;                              // 0..2047
    int g = r >> 9, h = r & 511;
    const float *wi, *wr, *wm;
    if (g == 0)      { wi = wi0; wr = wr0; wm = nullptr; }
    else if (g == 1) { wi = wi1; wr = wr1; wm = wm1; }
    else if (g == 2) { wi = wi2; wr = wr2; wm = wm2; }
    else             { wi = wi3; wr = wr3; wm = wm3; }
    int seg = k >> 9, kk = k & 511;
    float v;
    if (seg == 0)      v = wi[h * 513 + kk];
    else if (seg == 1) v = wr[h * 513 + kk];
    else               v = wm ? wm[h * 512 + kk] : 0.0f;
    Wp[(size_t)r * KTOT + k] = (__bf16)v;
    if (k == 0) bias[r] = wi[h * 513 + 512] + wr[h * 513 + 512];
}

// ---------------------------------------------------------------------------
// Kernel 2: build XT[b][k] bf16 (k contiguous) from f32 [k][b] sources.
// seg 0=input, 1=out0, 2=mem0. LDS-tiled 32x32 transpose.
// ---------------------------------------------------------------------------
__global__ void transpose_x(const float* __restrict__ input,
                            const float* __restrict__ out0,
                            const float* __restrict__ mem0,
                            __bf16* __restrict__ XT)
{
    __shared__ float tile[32][33];
    int tx = threadIdx.x & 31, ty = threadIdx.x >> 5;   // ty 0..7
    int b0 = blockIdx.x * 32, k0 = blockIdx.y * 32, seg = blockIdx.z;
    const float* src = (seg == 0) ? input : ((seg == 1) ? out0 : mem0);
#pragma unroll
    for (int i = 0; i < 4; ++i)
        tile[ty + 8 * i][tx] = src[(size_t)(k0 + ty + 8 * i) * B_DIM + b0 + tx];
    __syncthreads();
#pragma unroll
    for (int i = 0; i < 4; ++i) {
        int bb = ty + 8 * i;
        XT[(size_t)(b0 + bb) * KTOT + seg * 512 + k0 + tx] = (__bf16)tile[tx][bb];
    }
}

// ---------------------------------------------------------------------------
// Fragment load: two ds_load_b128 per 16x32 bf16 fragment
// ---------------------------------------------------------------------------
__device__ inline v16bf load_frag(const unsigned char* p, int off2)
{
    v16bf r;
    uint4* q = reinterpret_cast<uint4*>(&r);
    q[0] = *reinterpret_cast<const uint4*>(p);
    q[1] = *reinterpret_cast<const uint4*>(p + off2);
    return r;
}

__device__ inline float sigm(float x) { return 1.0f / (1.0f + __expf(-x)); }

// ---------------------------------------------------------------------------
// Kernel 3: fused GEMM (4 gates) + LSTM-cell epilogue.
// grid (B/128, H/64), 256 threads = 8 wave32 (wm 0..3 x wn 0..1).
// Per wave, per gate: 16 (M) x 64 (N) via 4 accumulators -> 16 WMMA / k-step.
// TDM double-buffered staging, wave0 drives tensor loads + tensorcnt waits.
// ---------------------------------------------------------------------------
__global__ __launch_bounds__(256, 1) void lstm_fused(
    const __bf16* __restrict__ XT, const __bf16* __restrict__ Wp,
    const float* __restrict__ bias, const float* __restrict__ mem0,
    float* __restrict__ out, float* __restrict__ memo)
{
    __shared__ __align__(128) unsigned char smem[2 * BUF_BYTES];

    const int tid  = threadIdx.x;
    const int wid  = tid >> 5;
    const int lane = tid & 31;
    const int hi   = lane >> 4;      // half-wave select (K-chunk)
    const int lrow = lane & 15;      // fragment row/col
    const int wm   = wid & 3;        // 16-row m sub-tile within 64-row gate tile
    const int wn   = wid >> 2;       // 64-col n half of 128-col tile

    const int h0 = blockIdx.y * 64;
    const int n0 = blockIdx.x * 128;

    const unsigned ldsbase = (unsigned)(unsigned long long)(uintptr_t)(void*)smem;

    // X: 2D tile 32(k) x 128(b) of XT[B][1536]; W: 3D tile 32(k) x 64(h) x 4(gate)
    const i32x8 g1X = make_g1(KTOT, B_DIM, 32, X_ROWS, 0, KTOT, 0);
    const i32x8 g1W = make_g1(KTOT, 512,   32, 64,     4, KTOT, 512ull * KTOT);
    const i32x4 g2X = {0, 0, 0, 0};
    const i32x4 g2W = {4, 0, 0, 0};   // tensor_dim2 = 4 gates

    unsigned long long gaX = (unsigned long long)(uintptr_t)XT +
                             (unsigned long long)n0 * (KTOT * 2);
    unsigned long long gaW = (unsigned long long)(uintptr_t)Wp +
                             (unsigned long long)h0 * (KTOT * 2);

    if (wid == 0) {  // prefetch buffers 0 and 1
        tdm_load(ldsbase,                       gaX,       g1X, g2X);
        tdm_load(ldsbase + X_BYTES,             gaW,       g1W, g2W);
        tdm_load(ldsbase + BUF_BYTES,           gaX + 64,  g1X, g2X);
        tdm_load(ldsbase + BUF_BYTES + X_BYTES, gaW + 64,  g1W, g2W);
    }

    v8f acc[4][4];
#pragma unroll
    for (int g = 0; g < 4; ++g)
#pragma unroll
        for (int j = 0; j < 4; ++j)
#pragma unroll
            for (int r = 0; r < 8; ++r) acc[g][j][r] = 0.0f;

    for (int kk = 0; kk < NK; ++kk) {
        if (wid == 0) {
            if (kk + 1 < NK) __builtin_amdgcn_s_wait_tensorcnt(2);
            else             __builtin_amdgcn_s_wait_tensorcnt(0);
        }
        __syncthreads();   // buffer (kk&1) published to all waves

        const unsigned char* bufp = smem + (kk & 1) * BUF_BYTES;
        const unsigned char* Xs = bufp;
        const unsigned char* Ws = bufp + X_BYTES;

        // B fragments: lane holds N=lrow, 16 consecutive k at byte 32*hi
        v16bf bf[4];
#pragma unroll
        for (int j = 0; j < 4; ++j) {
            const unsigned char* p =
                Xs + (unsigned)(wn * 64 + j * 16 + lrow) * ROWB + hi * 32;
            bf[j] = load_frag(p, 16);
        }
        // A fragments: lane holds M=lrow, 16 consecutive k at byte 32*hi
        v16bf af[4];
#pragma unroll
        for (int g = 0; g < 4; ++g) {
            const unsigned char* p =
                Ws + (unsigned)(g * 64 + wm * 16 + lrow) * ROWB + hi * 32;
            af[g] = load_frag(p, 16);
        }

#pragma unroll
        for (int g = 0; g < 4; ++g)
#pragma unroll
            for (int j = 0; j < 4; ++j)
                acc[g][j] = __builtin_amdgcn_wmma_f32_16x16x32_bf16(
                    false, af[g], false, bf[j], (short)0, acc[g][j], false, false);

        __syncthreads();   // everyone done reading buffer (kk&1)

        if (wid == 0 && kk + 2 < NK) {
            unsigned long long off = (unsigned long long)(kk + 2) * 64ull;
            unsigned lb = ldsbase + (unsigned)((kk & 1) * BUF_BYTES);
            tdm_load(lb,           gaX + off, g1X, g2X);
            tdm_load(lb + X_BYTES, gaW + off, g1W, g2W);
        }
    }

    // ---- fused LSTM-cell epilogue -----------------------------------------
    // C/D layout: lanes 0-15 -> N=lane, M=r ; lanes 16-31 -> N=lane-16, M=8+r
    const int hbase = h0 + wm * 16 + hi * 8;
    const int bbase = n0 + wn * 64 + lrow;

    float bsv[4][8];
#pragma unroll
    for (int g = 0; g < 4; ++g)
#pragma unroll
        for (int r = 0; r < 8; ++r)
            bsv[g][r] = bias[g * 512 + hbase + r];

#pragma unroll
    for (int j = 0; j < 4; ++j) {
        const int b = bbase + j * 16;
#pragma unroll
        for (int r = 0; r < 8; ++r) {
            const size_t idx = (size_t)(hbase + r) * B_DIM + b;
            float pi = acc[0][j][r] + bsv[0][r];
            float pg = acc[1][j][r] + bsv[1][r];
            float pr = acc[2][j][r] + bsv[2][r];
            float po = acc[3][j][r] + bsv[3][r];
            float t  = tanhf(pi);
            float gi = sigm(pg);
            float gr = sigm(pr);
            float go = sigm(po);
            float m0 = mem0[idx];
            float mv = t * gi + gr * m0;
            out[idx]  = go * mv;
            memo[idx] = mv;
        }
    }
}

// ---------------------------------------------------------------------------
// Host launcher
// ---------------------------------------------------------------------------
extern "C" void kernel_launch(void* const* d_in, const int* in_sizes, int n_in,
                              void* d_out, int out_size, void* d_ws, size_t ws_size,
                              hipStream_t stream)
{
    (void)in_sizes; (void)n_in; (void)out_size; (void)ws_size;

    const float* input = (const float*)d_in[0];
    const float* out0  = (const float*)d_in[1];
    const float* mem0  = (const float*)d_in[2];
    const float* w_ig  = (const float*)d_in[3];
    const float* wr_ig = (const float*)d_in[4];
    const float* wm_ig = (const float*)d_in[5];
    const float* w_in  = (const float*)d_in[6];
    const float* wr_in = (const float*)d_in[7];
    const float* w_rg  = (const float*)d_in[8];
    const float* wr_rg = (const float*)d_in[9];
    const float* wm_rg = (const float*)d_in[10];
    const float* w_og  = (const float*)d_in[11];
    const float* wr_og = (const float*)d_in[12];
    const float* wm_og = (const float*)d_in[13];

    char* ws = (char*)d_ws;
    __bf16* XT  = (__bf16*)ws;                                          // 96 MB
    __bf16* Wp  = (__bf16*)(ws + (size_t)B_DIM * KTOT * 2);             // 6 MB
    float*  bias = (float*)(ws + (size_t)B_DIM * KTOT * 2
                               + (size_t)2048 * KTOT * 2);              // 8 KB

    float* out  = (float*)d_out;
    float* memo = out + (size_t)H_DIM * B_DIM;

    pack_weights<<<dim3(KTOT / 256, 2048), 256, 0, stream>>>(
        w_in, wr_in, w_ig, wr_ig, wm_ig, w_rg, wr_rg, wm_rg,
        w_og, wr_og, wm_og, Wp, bias);

    transpose_x<<<dim3(B_DIM / 32, H_DIM / 32, 3), 256, 0, stream>>>(
        input, out0, mem0, XT);

    lstm_fused<<<dim3(B_DIM / 128, H_DIM / 64), 256, 0, stream>>>(
        XT, Wp, bias, mem0, out, memo);
}
